// NonLocalBlock_49606872268840
// MI455X (gfx1250) — matrix operand order repaired and checked by reading, not compiled, not run on hardware
//
#include <hip/hip_runtime.h>
#include <hip/hip_bf16.h>

#define CDIM 256
#define IDIM 128

typedef __attribute__((ext_vector_type(16))) _Float16 v16h;
typedef __attribute__((ext_vector_type(8)))  _Float16 v8h;
typedef __attribute__((ext_vector_type(8)))  float    v8f;

__device__ __forceinline__ v8f vzero8() {
    v8f z;
#pragma unroll
    for (int j = 0; j < 8; ++j) z[j] = 0.0f;
    return z;
}

__device__ __forceinline__ v8f wmma16(v16h a, v16h b, v8f c) {
    // D = A(16x32 f16) * B(32x16 f16) + C(16x16 f32)
    return __builtin_amdgcn_wmma_f32_16x16x32_f16(
        /*neg_a=*/false, a, /*neg_b=*/false, b,
        /*c_mod=*/(short)0, c, /*reuse_a=*/false, /*reuse_b=*/false);
}

// A-fragment (16x32, f16), row-major source with leading dim `ld` (elements).
// lanes 0-15: row=row0+lane, K = k0..k0+15 (contiguous)
// lanes 16-31: row=row0+lane-16, K = k0+16..k0+31
__device__ __forceinline__ v16h load_a_frag(const _Float16* __restrict__ base,
                                            int ld, int row0, int k0) {
    int lane = threadIdx.x & 31;
    int hi   = lane >> 4;
    const _Float16* p = base + (size_t)(row0 + (lane & 15)) * ld + k0 + hi * 16;
    v8h lo = *reinterpret_cast<const v8h*>(p);
    v8h hp = *reinterpret_cast<const v8h*>(p + 8);
    v16h out;
#pragma unroll
    for (int j = 0; j < 8; ++j) { out[j] = lo[j]; out[j + 8] = hp[j]; }
    return out;
}

// B-fragment (32x16, f16) from a K-contiguous store: element (K=k, N=col) at
// base[col*ldk + k].  lanes 0-15: col=col0+lane, K=k0..k0+15 (contiguous);
// lanes 16-31: col=col0+lane-16, K=k0+16..k0+31.
__device__ __forceinline__ v16h load_b_frag(const _Float16* __restrict__ base,
                                            int ldk, int col0, int k0) {
    int lane = threadIdx.x & 31;
    int k    = k0 + ((lane >> 4) << 4);
    return *reinterpret_cast<const v16h*>(base + (size_t)(col0 + (lane & 15)) * ldk + k);
}

// 16 contiguous f16 from LDS (two ds_load_b128)
__device__ __forceinline__ v16h load_frag_lds16(const _Float16* p) {
    v8h lo = *reinterpret_cast<const v8h*>(p);
    v8h hp = *reinterpret_cast<const v8h*>(p + 8);
    v16h out;
#pragma unroll
    for (int j = 0; j < 8; ++j) { out[j] = lo[j]; out[j + 8] = hp[j]; }
    return out;
}

// ---------------- pack kernels ----------------

// x [B,C,N] f32 -> xh [B,N,C] f16 (transpose + convert)
__global__ void pack_x_kernel(const float* __restrict__ x, _Float16* __restrict__ xh,
                              int Bb, int Nn) {
    size_t idx = (size_t)blockIdx.x * blockDim.x + threadIdx.x;
    size_t total = (size_t)Bb * CDIM * Nn;
    if (idx >= total) return;
    int n = (int)(idx % Nn);
    size_t t = idx / Nn;
    int c = (int)(t % CDIM);
    int b = (int)(t / CDIM);
    xh[((size_t)b * Nn + n) * CDIM + c] = (_Float16)x[idx];
}

__global__ void cvt_kernel(const float* __restrict__ src, _Float16* __restrict__ dst, int n) {
    int i = blockIdx.x * blockDim.x + threadIdx.x;
    if (i < n) dst[i] = (_Float16)src[i];
}

// ---------------- projection GEMMs ----------------
// out[n,i] = sum_c xh[b,n,c] * w[i,c] + bias[i]
// z==0 -> theta (qh [N,I]); z==1 -> phi (kh [N,I] == phi^T); z==2 -> g (gB [I,N])
__global__ __launch_bounds__(128) void proj_kernel(
    const _Float16* __restrict__ xh,
    const _Float16* __restrict__ wth, const _Float16* __restrict__ wph,
    const _Float16* __restrict__ wg,
    const float* __restrict__ bth, const float* __restrict__ bph,
    const float* __restrict__ bg,
    _Float16* __restrict__ qh, _Float16* __restrict__ kh,
    _Float16* __restrict__ gB, int Nn) {
    int z    = blockIdx.z;
    int b    = blockIdx.y;
    int wv   = threadIdx.x >> 5;
    int lane = threadIdx.x & 31;
    int hi   = lane >> 4;
    int n0   = blockIdx.x * 64 + wv * 16;

    const _Float16* w    = (z == 0) ? wth : (z == 1) ? wph : wg;
    const float*    bias = (z == 0) ? bth : (z == 1) ? bph : bg;
    const _Float16* xb   = xh + (size_t)b * Nn * CDIM;

    v8f acc[8];
#pragma unroll
    for (int t = 0; t < 8; ++t) acc[t] = vzero8();

    for (int ks = 0; ks < CDIM / 32; ++ks) {
        v16h a = load_a_frag(xb, CDIM, n0, ks * 32);
#pragma unroll
        for (int it = 0; it < 8; ++it) {
            v16h bf = load_b_frag(w, CDIM, it * 16, ks * 32);
            acc[it] = wmma16(a, bf, acc[it]);
        }
    }

#pragma unroll
    for (int it = 0; it < 8; ++it) {
        int i = it * 16 + (lane & 15);
        float bi = bias[i];
#pragma unroll
        for (int r = 0; r < 8; ++r) {
            int n = n0 + r + 8 * hi;
            float val = acc[it][r] + bi;
            if (z <= 1) {
                _Float16* o = (z == 0) ? qh : kh;
                o[((size_t)b * Nn + n) * IDIM + i] = (_Float16)val;
            } else {
                gB[((size_t)b * IDIM + i) * Nn + n] = (_Float16)val;
            }
        }
    }
}

// ---------------- fused flash attention ----------------
// Double-buffered async K/V staging to LDS (ASYNCcnt), WMMA from LDS fragments.
// Q=theta [N,I], K=phi^T [N,I], V=g [I,N].  Per-WG tile: 64 q-rows x 32 keys.

// Issue one 32-key tile copy: K-tile 8KB (contiguous) -> ksh[m][i];
// V-tile 8KB (32 f16 per row i, row stride N) -> vsh[i][m].
__device__ __forceinline__ void issue_kv_tile(const _Float16* kk, const _Float16* vv,
                                              _Float16* kshp, _Float16* vshp,
                                              int mb, int Nn) {
    int t = threadIdx.x;
    const char* kg = (const char*)(kk + (size_t)mb * IDIM);
    const char* vg = (const char*)vv + (size_t)mb * 2;
    uint32_t kbase = (uint32_t)(uintptr_t)kshp;
    uint32_t vbase = (uint32_t)(uintptr_t)vshp;
#pragma unroll
    for (int j = 0; j < 4; ++j) {
        uint32_t o = (uint32_t)(j * 128 + t) * 16u;
        uint64_t kga = (uint64_t)(uintptr_t)(kg + o);
        asm volatile("global_load_async_to_lds_b128 %0, %1, off"
                     :: "v"(kbase + o), "v"(kga) : "memory");
        uint32_t row   = o >> 6;     // 64B per V row (32 f16)
        uint32_t inrow = o & 63u;
        uint64_t vga = (uint64_t)(uintptr_t)(vg + (size_t)row * ((size_t)Nn * 2) + inrow);
        asm volatile("global_load_async_to_lds_b128 %0, %1, off"
                     :: "v"(vbase + o), "v"(vga) : "memory");
    }
}

__global__ __launch_bounds__(128) void attn_kernel(
    const _Float16* __restrict__ qh, const _Float16* __restrict__ kh,
    const _Float16* __restrict__ gB, _Float16* __restrict__ yh, int Nn) {
    __shared__ __align__(32) _Float16 ksh[2][32 * 128];   // [m][i], 8KB each
    __shared__ __align__(32) _Float16 vsh[2][128 * 32];   // [i][m], 8KB each
    __shared__ __align__(32) _Float16 psh[4][16 * 32];    // per-wave P transpose

    int b    = blockIdx.y;
    int wv   = threadIdx.x >> 5;
    int lane = threadIdx.x & 31;
    int hi   = lane >> 4;
    int n0   = blockIdx.x * 64 + wv * 16;

    const _Float16* q  = qh + (size_t)b * Nn * IDIM;
    const _Float16* kk = kh + (size_t)b * Nn * IDIM;
    const _Float16* vv = gB + (size_t)b * IDIM * Nn;

    v16h qf[4];
#pragma unroll
    for (int ks = 0; ks < 4; ++ks) qf[ks] = load_a_frag(q, IDIM, n0, ks * 32);

    v8f o[8];
#pragma unroll
    for (int t = 0; t < 8; ++t) o[t] = vzero8();
    float mrow[8], lrow[8];
#pragma unroll
    for (int r = 0; r < 8; ++r) { mrow[r] = -1e30f; lrow[r] = 0.0f; }

    issue_kv_tile(kk, vv, &ksh[0][0], &vsh[0][0], 0, Nn);

    for (int mt = 0; mt < Nn / 32; ++mt) {
        int p  = mt & 1;
        int mb = mt * 32;

        asm volatile("s_wait_asynccnt 0x0" ::: "memory");
        __syncthreads();
        if (mb + 32 < Nn)
            issue_kv_tile(kk, vv, &ksh[p ^ 1][0], &vsh[p ^ 1][0], mb + 32, Nn);

        // S = Q K^T for 32 keys (two 16-col tiles), K-dim = IDIM, from LDS
        v8f s0 = vzero8(), s1 = vzero8();
#pragma unroll
        for (int ks = 0; ks < 4; ++ks) {
            v16h b0 = load_frag_lds16(&ksh[p][((lane & 15)) * 128 + ks * 32 + hi * 16]);
            s0 = wmma16(qf[ks], b0, s0);
        }
#pragma unroll
        for (int ks = 0; ks < 4; ++ks) {
            v16h b1 = load_frag_lds16(&ksh[p][(16 + (lane & 15)) * 128 + ks * 32 + hi * 16]);
            s1 = wmma16(qf[ks], b1, s1);
        }

        // online softmax: row stats via 16-lane butterflies (rows stay in-half)
#pragma unroll
        for (int r = 0; r < 8; ++r) {
            float tm = fmaxf(s0[r], s1[r]);
#pragma unroll
            for (int msk = 1; msk < 16; msk <<= 1) tm = fmaxf(tm, __shfl_xor(tm, msk, 32));
            float mn = fmaxf(mrow[r], tm);
            float sc = __expf(mrow[r] - mn);
            float p0 = __expf(s0[r] - mn);
            float p1 = __expf(s1[r] - mn);
            float rs = p0 + p1;
#pragma unroll
            for (int msk = 1; msk < 16; msk <<= 1) rs += __shfl_xor(rs, msk, 32);
            lrow[r] = lrow[r] * sc + rs;
            mrow[r] = mn;
            s0[r] = p0; s1[r] = p1;
#pragma unroll
            for (int t = 0; t < 8; ++t) o[t][r] *= sc;
        }

        // P (C-layout f32) -> A-layout f16 via per-wave LDS transpose
#pragma unroll
        for (int r = 0; r < 8; ++r) {
            int row = r + 8 * hi;
            psh[wv][row * 32 + (lane & 15)]      = (_Float16)s0[r];
            psh[wv][row * 32 + 16 + (lane & 15)] = (_Float16)s1[r];
        }
        asm volatile("s_wait_dscnt 0x0" ::: "memory");
        v16h pa;
        {
            const _Float16* pp = &psh[wv][(lane & 15) * 32 + hi * 16];
            v8h lo = *reinterpret_cast<const v8h*>(pp);
            v8h hp = *reinterpret_cast<const v8h*>(pp + 8);
#pragma unroll
            for (int j = 0; j < 8; ++j) { pa[j] = lo[j]; pa[j + 8] = hp[j]; }
        }

        // O += P . V   (V fragment from vsh[i][m]: K=m contiguous)
#pragma unroll
        for (int it = 0; it < 8; ++it) {
            v16h vb = load_frag_lds16(&vsh[p][(it * 16 + (lane & 15)) * 32 + hi * 16]);
            o[it] = wmma16(pa, vb, o[it]);
        }
    }

    // normalize and store y [N, I] f16
#pragma unroll
    for (int it = 0; it < 8; ++it) {
        int i = it * 16 + (lane & 15);
#pragma unroll
        for (int r = 0; r < 8; ++r) {
            int n = n0 + r + 8 * hi;
            float val = o[it][r] / lrow[r];
            yh[((size_t)b * Nn + n) * IDIM + i] = (_Float16)val;
        }
    }
}

// ---------------- output projection + residual ----------------
// out[b,c,n] = sum_i y[n,i]*w_W[c,i] + b_W[c] + x[b,c,n]
__global__ __launch_bounds__(128) void outproj_kernel(
    const _Float16* __restrict__ yh, const _Float16* __restrict__ wW,
    const float* __restrict__ bW, const float* __restrict__ x,
    float* __restrict__ out, int Nn) {
    int b    = blockIdx.y;
    int wv   = threadIdx.x >> 5;
    int lane = threadIdx.x & 31;
    int hi   = lane >> 4;
    int n0   = blockIdx.x * 64 + wv * 16;

    const _Float16* yb = yh + (size_t)b * Nn * IDIM;

    v8f acc[16];
#pragma unroll
    for (int t = 0; t < 16; ++t) acc[t] = vzero8();

#pragma unroll
    for (int ks = 0; ks < IDIM / 32; ++ks) {
        v16h a = load_a_frag(yb, IDIM, n0, ks * 32);
#pragma unroll
        for (int ct = 0; ct < 16; ++ct) {
            v16h bf = load_b_frag(wW, IDIM, ct * 16, ks * 32);
            acc[ct] = wmma16(a, bf, acc[ct]);
        }
    }

#pragma unroll
    for (int ct = 0; ct < 16; ++ct) {
        int c = ct * 16 + (lane & 15);
        float bc = bW[c];
#pragma unroll
        for (int r = 0; r < 8; ++r) {
            int n = n0 + r + 8 * hi;
            size_t idx = ((size_t)b * CDIM + c) * Nn + n;
            out[idx] = acc[ct][r] + bc + x[idx];
        }
    }
}

extern "C" void kernel_launch(void* const* d_in, const int* in_sizes, int n_in,
                              void* d_out, int out_size, void* d_ws, size_t ws_size,
                              hipStream_t stream) {
    const float* x    = (const float*)d_in[0];
    const float* w_g  = (const float*)d_in[1];
    const float* b_g  = (const float*)d_in[2];
    const float* w_th = (const float*)d_in[3];
    const float* b_th = (const float*)d_in[4];
    const float* w_ph = (const float*)d_in[5];
    const float* b_ph = (const float*)d_in[6];
    const float* w_W  = (const float*)d_in[7];
    const float* b_W  = (const float*)d_in[8];

    const int Nn = 64 * 64;
    const int Bb = in_sizes[0] / (CDIM * Nn);

    char* ws = (char*)d_ws;
    size_t off = 0;
    auto take = [&](size_t bytes) -> char* {
        char* p = ws + off;
        off = (off + bytes + 255) & ~(size_t)255;
        return p;
    };
    _Float16* xh   = (_Float16*)take((size_t)Bb * Nn * CDIM * 2);
    _Float16* qh   = (_Float16*)take((size_t)Bb * Nn * IDIM * 2);
    _Float16* kh   = (_Float16*)take((size_t)Bb * Nn * IDIM * 2);
    _Float16* gB   = (_Float16*)take((size_t)Bb * IDIM * Nn * 2);
    _Float16* yh   = (_Float16*)take((size_t)Bb * Nn * IDIM * 2);
    _Float16* wgh  = (_Float16*)take((size_t)IDIM * CDIM * 2);
    _Float16* wthh = (_Float16*)take((size_t)IDIM * CDIM * 2);
    _Float16* wphh = (_Float16*)take((size_t)IDIM * CDIM * 2);
    _Float16* wWh  = (_Float16*)take((size_t)CDIM * IDIM * 2);

    {
        size_t total = (size_t)Bb * CDIM * Nn;
        int blocks = (int)((total + 255) / 256);
        pack_x_kernel<<<blocks, 256, 0, stream>>>(x, xh, Bb, Nn);
    }
    {
        int nw = IDIM * CDIM;
        cvt_kernel<<<(nw + 255) / 256, 256, 0, stream>>>(w_g, wgh, nw);
        cvt_kernel<<<(nw + 255) / 256, 256, 0, stream>>>(w_th, wthh, nw);
        cvt_kernel<<<(nw + 255) / 256, 256, 0, stream>>>(w_ph, wphh, nw);
        cvt_kernel<<<(nw + 255) / 256, 256, 0, stream>>>(w_W, wWh, nw);
    }

    dim3 blk(128, 1, 1);
    proj_kernel<<<dim3(Nn / 64, Bb, 3), blk, 0, stream>>>(
        xh, wthh, wphh, wgh, b_th, b_ph, b_g, qh, kh, gB, Nn);
    attn_kernel<<<dim3(Nn / 64, Bb), blk, 0, stream>>>(qh, kh, gB, yh, Nn);
    outproj_kernel<<<dim3(Nn / 64, Bb), blk, 0, stream>>>(yh, wWh, b_W, x, (float*)d_out, Nn);
}
